// WanSelfAttention_81046032876244
// MI455X (gfx1250) — compile-verified, matrix operand order checked by reading, not run on hardware
//
#include <hip/hip_runtime.h>
#include <hip/hip_bf16.h>

typedef __attribute__((ext_vector_type(16))) _Float16 v16h;
typedef __attribute__((ext_vector_type(8)))  float    v8f;

#define DMODEL 1536
#define NHEADS 12
#define HDIM   128
#define CPAIR  64           // rope complex channels per head
#define LTOK   4096
#define S1C    22           // frame-axis channels
#define S2C    21           // h-axis channels (w gets the remaining 21)
#define QSCALE 0.08838834764831845f   // 1/sqrt(128)

// ---------------------------------------------------------------------------
// 16-bit WMMA fragment loader (A or B) from a row-major matrix.
// A-matrix 16x32 layout (ISA 7.12.2): lane l holds row (row_base + l%16);
// halves h=0..7 -> K = kg..kg+7, h=8..15 -> K = kg+16..kg+23, kg += 8 for
// lanes 16..31.  B-matrix (KxN) mirrors this with lane = column index, so the
// same gather works when 'src' is the N-major (i.e. W row-major / K row-major /
// V-transposed) matrix.  Gathers vectorize to two global_load_b128 per frag.
// ---------------------------------------------------------------------------
__device__ __forceinline__ v16h load_frag16(const _Float16* __restrict__ src,
                                            int ld, int row_base, int k_base) {
  const int lane = threadIdx.x & 31;
  const _Float16* p = src + (size_t)(row_base + (lane & 15)) * ld
                          + k_base + ((lane >> 4) << 3);
  v16h f;
#pragma unroll
  for (int h = 0; h < 8; ++h) { f[h] = p[h]; f[h + 8] = p[16 + h]; }
  return f;
}

__device__ __forceinline__ float redmax16(float v) {
#pragma unroll
  for (int off = 8; off; off >>= 1) v = fmaxf(v, __shfl_xor(v, off, 32));
  return v;
}
__device__ __forceinline__ float redsum16(float v) {
#pragma unroll
  for (int off = 8; off; off >>= 1) v += __shfl_xor(v, off, 32);
  return v;
}

// ---------------------------------------------------------------------------
// elementwise fp32 -> fp16 convert
// ---------------------------------------------------------------------------
__global__ void f32_to_f16(const float* __restrict__ in,
                           _Float16* __restrict__ out, int n) {
  int i = blockIdx.x * 256 + threadIdx.x;
  if (i < n) out[i] = (_Float16)in[i];
}

// ---------------------------------------------------------------------------
// C[M,N] = A[M,K](f16) @ W[N,K](f16)^T + bias   (fp32 out)
// block = 256 (8 waves); block tile 256x64; wave tile 32x64:
// 2 A-frags x 4 B-frags = 8 WMMAs per K-step (1.5 b128 loads per wmma).
// ---------------------------------------------------------------------------
__global__ __launch_bounds__(256)
void gemm_xwT_bias(const _Float16* __restrict__ A, const _Float16* __restrict__ W,
                   const float* __restrict__ bias, float* __restrict__ C,
                   int M, int N, int K) {
  const int wave = threadIdx.x >> 5;
  const int lane = threadIdx.x & 31;
  const int mBase = blockIdx.y * 256 + wave * 32;
  const int nBase = blockIdx.x * 64;

  const v8f vzero = {0.f,0.f,0.f,0.f,0.f,0.f,0.f,0.f};
  v8f acc0[4], acc1[4];
#pragma unroll
  for (int t = 0; t < 4; ++t) { acc0[t] = vzero; acc1[t] = vzero; }

  for (int k = 0; k < K; k += 32) {
    // warm L0/WGP$ for the next K-step (gfx1250 global_prefetch_b8)
    if (k + 32 < K) {
      __builtin_prefetch(A + (size_t)(mBase + (lane & 15)) * K + k + 32, 0, 1);
      __builtin_prefetch(A + (size_t)(mBase + 16 + (lane & 15)) * K + k + 32, 0, 1);
      __builtin_prefetch(W + (size_t)(nBase + (lane & 15)) * K + k + 32, 0, 1);
      __builtin_prefetch(W + (size_t)(nBase + 32 + (lane & 15)) * K + k + 32, 0, 1);
    }
    v16h a0 = load_frag16(A, K, mBase,      k);
    v16h a1 = load_frag16(A, K, mBase + 16, k);
#pragma unroll
    for (int t = 0; t < 4; ++t) {
      v16h b = load_frag16(W, K, nBase + t * 16, k);
      acc0[t] = __builtin_amdgcn_wmma_f32_16x16x32_f16(
                  false, a0, false, b, (short)0, acc0[t], false, false);
      acc1[t] = __builtin_amdgcn_wmma_f32_16x16x32_f16(
                  false, a1, false, b, (short)0, acc1[t], false, false);
    }
  }

  const int g = lane >> 4, n = lane & 15;
#pragma unroll
  for (int t = 0; t < 4; ++t) {
    const int col = nBase + t * 16 + n;
    const float bvv = bias[col];
#pragma unroll
    for (int r = 0; r < 8; ++r) {
      C[(size_t)(mBase + r + 8 * g) * N + col]      = acc0[t][r] + bvv;
      C[(size_t)(mBase + 16 + r + 8 * g) * N + col] = acc1[t][r] + bvv;
    }
  }
}

// ---------------------------------------------------------------------------
// Fused RMSNorm (over full 1536) + 3D RoPE, fp32 -> f16.
// one block (256 threads) per token row.  out_scale folds 1/sqrt(d) into Q.
// ---------------------------------------------------------------------------
__global__ __launch_bounds__(256)
void rms_rope_kernel(const float* __restrict__ y, const float* __restrict__ gain,
                     const float* __restrict__ fcos, const float* __restrict__ fsin,
                     const int* __restrict__ fsz, const int* __restrict__ hsz,
                     const int* __restrict__ wsz,
                     _Float16* __restrict__ out, float out_scale) {
  const int row = blockIdx.x;
  const float* yr = y + (size_t)row * DMODEL;
  __shared__ float red[256];

  float ss = 0.f;
  for (int i = threadIdx.x; i < DMODEL; i += 256) { float v = yr[i]; ss += v * v; }
  red[threadIdx.x] = ss;
  __syncthreads();
#pragma unroll
  for (int s = 128; s > 0; s >>= 1) {
    if (threadIdx.x < s) red[threadIdx.x] += red[threadIdx.x + s];
    __syncthreads();
  }
  const float inv = rsqrtf(red[0] * (1.0f / DMODEL) + 1e-6f);

  const int Hs = *hsz, Ws = *wsz; (void)fsz;
  const int fpos = row / (Hs * Ws);
  const int rem  = row % (Hs * Ws);
  const int hpos = rem / Ws;
  const int wpos = rem % Ws;

  for (int p = threadIdx.x; p < DMODEL / 2; p += 256) {
    const int c = p & (CPAIR - 1);                  // channel within head
    const int pos = (c < S1C) ? fpos : ((c < S1C + S2C) ? hpos : wpos);
    const float cs = fcos[pos * CPAIR + c];
    const float sn = fsin[pos * CPAIR + c];
    const float xr = yr[2 * p]     * inv * gain[2 * p];
    const float xi = yr[2 * p + 1] * inv * gain[2 * p + 1];
    out[(size_t)row * DMODEL + 2 * p]     = (_Float16)((xr * cs - xi * sn) * out_scale);
    out[(size_t)row * DMODEL + 2 * p + 1] = (_Float16)((xr * sn + xi * cs) * out_scale);
  }
}

// ---------------------------------------------------------------------------
// V fp32 [L, D] -> per-head transposed f16 [head][128][L]
// ---------------------------------------------------------------------------
__global__ void v_transpose(const float* __restrict__ v, _Float16* __restrict__ vt) {
  int i = blockIdx.x * 256 + threadIdx.x;
  if (i >= LTOK * DMODEL) return;
  const int key = i / DMODEL;
  const int col = i % DMODEL;      // = head*128 + hc
  vt[(size_t)col * LTOK + key] = (_Float16)v[i];
}

// ---------------------------------------------------------------------------
// Flash attention: one wave per (head, 16-row Q block).
// Q,K f16 [L, D] (Q pre-scaled by 1/sqrt(d)), Vt f16 [head][128][L].
// ---------------------------------------------------------------------------
__global__ __launch_bounds__(32)
void attn_kernel(const _Float16* __restrict__ Q, const _Float16* __restrict__ Kb,
                 const _Float16* __restrict__ Vt, _Float16* __restrict__ O) {
  const int head = blockIdx.y;
  const int qb   = blockIdx.x * 16;
  const int lane = threadIdx.x;
  const int g = lane >> 4, n = lane & 15;

  __shared__ _Float16 lds_p[16 * 32];

  v16h qf[4];
#pragma unroll
  for (int i = 0; i < 4; ++i)
    qf[i] = load_frag16(Q, DMODEL, qb, head * HDIM + i * 32);

  const v8f vzero = {0.f,0.f,0.f,0.f,0.f,0.f,0.f,0.f};
  v8f out[8];
#pragma unroll
  for (int t = 0; t < 8; ++t) out[t] = vzero;
  float m_row[8], l_row[8];
#pragma unroll
  for (int r = 0; r < 8; ++r) { m_row[r] = -1e30f; l_row[r] = 0.f; }

  const _Float16* vh = Vt + (size_t)head * HDIM * LTOK;

  for (int kb = 0; kb < LTOK; kb += 32) {
    // prefetch next key block rows of K into near cache
    if (kb + 32 < LTOK) {
      __builtin_prefetch(Kb + (size_t)(kb + 32 + lane) * DMODEL + head * HDIM, 0, 1);
    }
    // ---- S = Q . K^T  (16 q-rows x 32 keys) ----
    v8f s0 = vzero, s1 = vzero;
#pragma unroll
    for (int i = 0; i < 4; ++i) {
      v16h b0 = load_frag16(Kb, DMODEL, kb,      head * HDIM + i * 32);
      v16h b1 = load_frag16(Kb, DMODEL, kb + 16, head * HDIM + i * 32);
      s0 = __builtin_amdgcn_wmma_f32_16x16x32_f16(false, qf[i], false, b0, (short)0, s0, false, false);
      s1 = __builtin_amdgcn_wmma_f32_16x16x32_f16(false, qf[i], false, b1, (short)0, s1, false, false);
    }

    // ---- online softmax (rows live as vgpr r within 16-lane group g) ----
#pragma unroll
    for (int r = 0; r < 8; ++r) {
      float mx = redmax16(fmaxf(s0[r], s1[r]));
      float mn = fmaxf(m_row[r], mx);
      float al = __expf(m_row[r] - mn);
      float p0 = __expf(s0[r] - mn);
      float p1 = __expf(s1[r] - mn);
      l_row[r] = l_row[r] * al + redsum16(p0 + p1);
      m_row[r] = mn;
      s0[r] = p0; s1[r] = p1;
#pragma unroll
      for (int t = 0; t < 8; ++t) out[t][r] *= al;
    }

    // ---- transpose P (D-layout) -> A-fragment via LDS ----
#pragma unroll
    for (int r = 0; r < 8; ++r) {
      const int m = r + 8 * g;
      lds_p[m * 32 + n]      = (_Float16)s0[r];
      lds_p[m * 32 + 16 + n] = (_Float16)s1[r];
    }
    __syncthreads();
    v16h pa;
    {
      const int m = lane & 15;
      const int kg = (lane >> 4) << 3;
#pragma unroll
      for (int h = 0; h < 8; ++h) {
        pa[h]     = lds_p[m * 32 + kg + h];
        pa[h + 8] = lds_p[m * 32 + kg + 16 + h];
      }
    }
    __syncthreads();

    // ---- O += P . V  (K = 32 keys) ----
#pragma unroll
    for (int t = 0; t < 8; ++t) {
      v16h bv = load_frag16(vh, LTOK, t * 16, kb);
      out[t] = __builtin_amdgcn_wmma_f32_16x16x32_f16(false, pa, false, bv, (short)0, out[t], false, false);
    }
  }

  // ---- normalize + store f16 for output projection ----
#pragma unroll
  for (int r = 0; r < 8; ++r) {
    const int m = qb + r + 8 * g;
    const float il = 1.0f / l_row[r];
#pragma unroll
    for (int t = 0; t < 8; ++t)
      O[(size_t)m * DMODEL + head * HDIM + t * 16 + n] = (_Float16)(out[t][r] * il);
  }
}

// ---------------------------------------------------------------------------
// launch
// ---------------------------------------------------------------------------
extern "C" void kernel_launch(void* const* d_in, const int* in_sizes, int n_in,
                              void* d_out, int out_size, void* d_ws, size_t ws_size,
                              hipStream_t stream) {
  (void)in_sizes; (void)n_in; (void)out_size; (void)ws_size;
  const float* x   = (const float*)d_in[0];
  const float* wq  = (const float*)d_in[1];
  const float* wk  = (const float*)d_in[2];
  const float* wv  = (const float*)d_in[3];
  const float* wo  = (const float*)d_in[4];
  const float* bq  = (const float*)d_in[5];
  const float* bk  = (const float*)d_in[6];
  const float* bv  = (const float*)d_in[7];
  const float* bo  = (const float*)d_in[8];
  const float* gq  = (const float*)d_in[9];
  const float* gk  = (const float*)d_in[10];
  const float* fc  = (const float*)d_in[11];
  const float* fs  = (const float*)d_in[12];
  const int* fsz   = (const int*)d_in[14];
  const int* hsz   = (const int*)d_in[15];
  const int* wsz   = (const int*)d_in[16];

  // workspace layout
  char* ws = (char*)d_ws;
  const size_t XB  = (size_t)LTOK * DMODEL * 2;   // f16 [L,D]
  const size_t WB  = (size_t)DMODEL * DMODEL * 2; // f16 [D,D]
  const size_t TMP = (size_t)LTOK * DMODEL * 4;   // f32 [L,D]
  _Float16* xb    = (_Float16*)ws;
  _Float16* wbuf  = (_Float16*)(ws + XB);
  float*    tmp   = (float*)   (ws + XB + WB);
  _Float16* qbuf  = (_Float16*)(ws + XB + WB + TMP);
  _Float16* kbuf  = (_Float16*)(ws + XB + WB + TMP + XB);
  _Float16* vtb   = (_Float16*)(ws + XB + WB + TMP + 2 * XB);
  _Float16* attnb = (_Float16*)(ws + XB + WB + TMP + 3 * XB);

  const int LD = LTOK * DMODEL, DD = DMODEL * DMODEL;
  const dim3 gemmGrid(DMODEL / 64, LTOK / 256);

  // x -> f16
  f32_to_f16<<<(LD + 255) / 256, 256, 0, stream>>>(x, xb, LD);

  // ---- Q ----
  f32_to_f16<<<(DD + 255) / 256, 256, 0, stream>>>(wq, wbuf, DD);
  gemm_xwT_bias<<<gemmGrid, 256, 0, stream>>>(xb, wbuf, bq, tmp, LTOK, DMODEL, DMODEL);
  rms_rope_kernel<<<LTOK, 256, 0, stream>>>(tmp, gq, fc, fs, fsz, hsz, wsz, qbuf, QSCALE);

  // ---- K ----
  f32_to_f16<<<(DD + 255) / 256, 256, 0, stream>>>(wk, wbuf, DD);
  gemm_xwT_bias<<<gemmGrid, 256, 0, stream>>>(xb, wbuf, bk, tmp, LTOK, DMODEL, DMODEL);
  rms_rope_kernel<<<LTOK, 256, 0, stream>>>(tmp, gk, fc, fs, fsz, hsz, wsz, kbuf, 1.0f);

  // ---- V ----
  f32_to_f16<<<(DD + 255) / 256, 256, 0, stream>>>(wv, wbuf, DD);
  gemm_xwT_bias<<<gemmGrid, 256, 0, stream>>>(xb, wbuf, bv, tmp, LTOK, DMODEL, DMODEL);
  v_transpose<<<(LD + 255) / 256, 256, 0, stream>>>(tmp, vtb);

  // ---- attention ----
  attn_kernel<<<dim3(LTOK / 16, NHEADS), 32, 0, stream>>>(qbuf, kbuf, vtb, attnb);

  // ---- output projection ----
  f32_to_f16<<<(DD + 255) / 256, 256, 0, stream>>>(wo, wbuf, DD);
  gemm_xwT_bias<<<gemmGrid, 256, 0, stream>>>(attnb, wbuf, bo, (float*)d_out,
                                              LTOK, DMODEL, DMODEL);
}
